// MultiHeadAttention_58136677318861
// MI455X (gfx1250) — matrix operand order repaired and checked by reading, not compiled, not run on hardware
//
#include <hip/hip_runtime.h>

// ---------------------------------------------------------------------------
// CDNA5 (gfx1250, wave32) fused MHA forward:
//   1) qkv  = x @ Wqkv^T          (WMMA bf16, f32 accum)
//   2) q,k  = rope(rmsnorm(qkv))  -> bf16 [B,H,L,D];  v -> bf16 [B,H,D,L]
//   3) o    = causal softmax(q k^T / sqrt(D)) v   (flash, WMMA bf16,
//             K/V staged with GLOBAL_LOAD_ASYNC_TO_LDS_B128)
//   4) out  = o @ Wout^T          (WMMA bf16, f32 accum)
// ---------------------------------------------------------------------------

typedef __attribute__((ext_vector_type(16))) __bf16 v16bf;
typedef __attribute__((ext_vector_type(8)))  __bf16 v8bf;
typedef __attribute__((ext_vector_type(8)))  float  v8f;

__device__ __forceinline__ __bf16 f2bf(float f) {
  unsigned u = __builtin_bit_cast(unsigned, f);
  u += 0x7fffu + ((u >> 16) & 1u);                  // round-to-nearest-even
  unsigned short h = (unsigned short)(u >> 16);
  return __builtin_bit_cast(__bf16, h);
}

__device__ __forceinline__ v16bf cat8(v8bf lo, v8bf hi) {
  return __builtin_shufflevector(lo, hi, 0, 1, 2, 3, 4, 5, 6, 7,
                                 8, 9, 10, 11, 12, 13, 14, 15);
}

__device__ __forceinline__ v8f wmma_bf16(v16bf a, v16bf b, v8f c) {
  // v_wmma_f32_16x16x32_bf16  D = A x B + C
  return __builtin_amdgcn_wmma_f32_16x16x32_bf16(false, a, false, b,
                                                 (short)0, c, false, false);
}

// Async LDS-DMA: copy 16 bytes global -> LDS without touching VGPR data.
// VDST = per-lane LDS byte address (low 32 bits of the generic shared ptr,
// per the flat-aperture mapping); tracked by ASYNCcnt.
__device__ __forceinline__ void async_copy_b128(void* lds_dst,
                                                const void* gsrc) {
  unsigned lds32 = (unsigned)(unsigned long long)lds_dst;
  asm volatile("global_load_async_to_lds_b128 %0, %1, off"
               :
               : "v"(lds32), "v"(gsrc)
               : "memory");
}

__device__ __forceinline__ void wait_async0() {
  asm volatile("s_wait_asynccnt 0" ::: "memory");
}

// ---------------------------------------------------------------------------
// Generic GEMM: C[M,N] f32 = A[M,K] (f32 or bf16) * B[N,K]^T (f32, rows = N)
// Block tile 64(M) x 128(N), K-step 32.  8 waves as 2x4 grid, each wave 32x32.
// ---------------------------------------------------------------------------
template <bool A_BF16>
__global__ __launch_bounds__(256) void gemm_bf16_wmma(
    const void* __restrict__ Av, const float* __restrict__ Bw,
    float* __restrict__ C, int M, int N, int K, int ldc) {
  (void)M; (void)N;
  constexpr int BK = 32;
  __shared__ __bf16 As[64][BK + 8];
  __shared__ __bf16 Bs[128][BK + 8];

  const int tid = threadIdx.x;
  const int wv = tid >> 5, lane = tid & 31;
  const int wm = wv >> 2, wn = wv & 3;          // 2 x 4 wave grid
  const int hl = lane >> 4, lr = lane & 15;     // half-wave, lane-in-half
  const int m0 = blockIdx.x * 64;
  const int n0 = blockIdx.y * 128;

  v8f acc[2][2] = {};

  for (int k0 = 0; k0 < K; k0 += BK) {
    // ---- stage A tile (64 x 32) ----
    {
      const int r = tid >> 2;
      const int c = (tid & 3) * 8;
      if constexpr (A_BF16) {
        const __bf16* ap = (const __bf16*)Av + (size_t)(m0 + r) * K + k0 + c;
        *(v8bf*)&As[r][c] = *(const v8bf*)ap;
      } else {
        const float* ap = (const float*)Av + (size_t)(m0 + r) * K + k0 + c;
        const float4 f0 = *(const float4*)ap;
        const float4 f1 = *(const float4*)(ap + 4);
        v8bf hv;
        hv[0] = f2bf(f0.x); hv[1] = f2bf(f0.y); hv[2] = f2bf(f0.z); hv[3] = f2bf(f0.w);
        hv[4] = f2bf(f1.x); hv[5] = f2bf(f1.y); hv[6] = f2bf(f1.z); hv[7] = f2bf(f1.w);
        *(v8bf*)&As[r][c] = hv;
      }
    }
    // ---- stage B tile (128 x 32): W rows are K-contiguous = B^T columns ----
    {
      const int r = tid >> 1;
      const int c = (tid & 1) * 16;
      const float* bp = Bw + (size_t)(n0 + r) * K + k0 + c;
      const float4 f0 = *(const float4*)bp;
      const float4 f1 = *(const float4*)(bp + 4);
      const float4 f2 = *(const float4*)(bp + 8);
      const float4 f3 = *(const float4*)(bp + 12);
      v8bf h0, h1;
      h0[0] = f2bf(f0.x); h0[1] = f2bf(f0.y); h0[2] = f2bf(f0.z); h0[3] = f2bf(f0.w);
      h0[4] = f2bf(f1.x); h0[5] = f2bf(f1.y); h0[6] = f2bf(f1.z); h0[7] = f2bf(f1.w);
      h1[0] = f2bf(f2.x); h1[1] = f2bf(f2.y); h1[2] = f2bf(f2.z); h1[3] = f2bf(f2.w);
      h1[4] = f2bf(f3.x); h1[5] = f2bf(f3.y); h1[6] = f2bf(f3.z); h1[7] = f2bf(f3.w);
      *(v8bf*)&Bs[r][c] = h0;
      *(v8bf*)&Bs[r][c + 8] = h1;
    }
    __syncthreads();

    // ---- fragments + 4 WMMA ----
    v16bf af[2], bfm[2];
#pragma unroll
    for (int mi = 0; mi < 2; ++mi) {
      // A 16x32 bf16: halves 0/1 hold K [0..7,16..23] / [8..15,24..31]
      const __bf16* p = &As[wm * 32 + mi * 16 + lr][hl * 8];
      af[mi] = cat8(*(const v8bf*)p, *(const v8bf*)(p + 16));
    }
#pragma unroll
    for (int ni = 0; ni < 2; ++ni) {
      // B 32x16 bf16: lane = column N, halves 0/1 hold K [0..7,16..23] / [8..15,24..31]
      const __bf16* p = &Bs[wn * 32 + ni * 16 + lr][hl * 8];
      bfm[ni] = cat8(*(const v8bf*)p, *(const v8bf*)(p + 16));
    }
#pragma unroll
    for (int mi = 0; mi < 2; ++mi)
#pragma unroll
      for (int ni = 0; ni < 2; ++ni)
        acc[mi][ni] = wmma_bf16(af[mi], bfm[ni], acc[mi][ni]);
    __syncthreads();
  }

  // ---- epilogue: D layout lane<16 -> M=j, lane>=16 -> M=j+8; N = lane&15 ----
#pragma unroll
  for (int mi = 0; mi < 2; ++mi)
#pragma unroll
    for (int ni = 0; ni < 2; ++ni) {
      const int row0 = m0 + wm * 32 + mi * 16 + hl * 8;
      const int col = n0 + wn * 32 + ni * 16 + lr;
#pragma unroll
      for (int j = 0; j < 8; ++j)
        C[(size_t)(row0 + j) * ldc + col] = acc[mi][ni][j];
    }
}

// ---------------------------------------------------------------------------
// RMSNorm + interleaved RoPE + bf16 repack.
// grid = B*L blocks, 256 threads; wave w handles heads w and w+8.
// ---------------------------------------------------------------------------
__global__ __launch_bounds__(256) void rmsnorm_rope_pack(
    const float* __restrict__ qkv, __bf16* __restrict__ qb,
    __bf16* __restrict__ kb, __bf16* __restrict__ vtb) {
  constexpr int L = 2048, Dh = 128, H = 16;
  const int bl = blockIdx.x;            // b*L + l
  const int l = bl & (L - 1);
  const int wv = threadIdx.x >> 5, lane = threadIdx.x & 31;
  const float* row = qkv + (size_t)bl * 6144;
  const float ln1e4 = 9.210340371976184f;   // ln(10000)

#pragma unroll
  for (int hh = 0; hh < 2; ++hh) {
    const int h = wv + hh * 8;
    const int bh = (bl >> 11) * H + h;      // b*H + h  (L = 2048 = 1<<11)
#pragma unroll
    for (int which = 0; which < 2; ++which) {   // 0 = q, 1 = k
      const float4 xv = *(const float4*)(row + which * 2048 + h * Dh + lane * 4);
      float ss = xv.x * xv.x + xv.y * xv.y + xv.z * xv.z + xv.w * xv.w;
#pragma unroll
      for (int m = 16; m > 0; m >>= 1) ss += __shfl_xor(ss, m, 32);
      const float scale = rsqrtf(ss * (1.0f / Dh) + 1e-5f);
      float o[4];
#pragma unroll
      for (int pi = 0; pi < 2; ++pi) {
        const int p = lane * 2 + pi;        // pair index
        const float xe = (pi ? xv.z : xv.x) * scale;   // x[2p]
        const float xo = (pi ? xv.w : xv.y) * scale;   // x[2p+1]
        const float inv = __expf(-((float)(2 * p) / (float)Dh) * ln1e4);
        const float ang = (float)l * inv;
        const float cs = cosf(ang), sn = sinf(ang);
        o[2 * pi + 0] = xo * cs - xe * sn;  // even slot = y_odd (ref quirk)
        o[2 * pi + 1] = xo * sn + xe * cs;  // odd slot  = y_even
      }
      __bf16* dst = (which ? kb : qb) + ((size_t)bh * L + l) * Dh + lane * 4;
      dst[0] = f2bf(o[0]); dst[1] = f2bf(o[1]);
      dst[2] = f2bf(o[2]); dst[3] = f2bf(o[3]);
    }
    {   // v: transpose to [B,H,D,L] so P*V B-fragments are K-contiguous
      const float4 vvv = *(const float4*)(row + 4096 + h * Dh + lane * 4);
      __bf16* base = vtb + (size_t)bh * Dh * L;
      const int d0 = lane * 4;
      base[(size_t)(d0 + 0) * L + l] = f2bf(vvv.x);
      base[(size_t)(d0 + 1) * L + l] = f2bf(vvv.y);
      base[(size_t)(d0 + 2) * L + l] = f2bf(vvv.z);
      base[(size_t)(d0 + 3) * L + l] = f2bf(vvv.w);
    }
  }
}

// ---------------------------------------------------------------------------
// Causal flash attention.  grid = (L/64, B*H), 128 threads (4 waves).
// Each wave owns 16 query rows; key tiles of 32; D = 128.
// K/V tiles staged via async LDS-DMA (no VGPR round trip).
// ---------------------------------------------------------------------------
__global__ __launch_bounds__(128) void attn_fwd_wmma(
    const __bf16* __restrict__ q, const __bf16* __restrict__ k,
    const __bf16* __restrict__ vt, __bf16* __restrict__ po) {
  constexpr int L = 2048, Dh = 128;
  __shared__ __bf16 Ks[32][136];       // [key][d]
  __shared__ __bf16 Vs[128][40];       // [d][key]
  __shared__ __bf16 Ps[4][16][40];     // per-wave P bounce [row][key]

  const int bh = blockIdx.y;           // b*16 + h
  const int b = bh >> 4, h = bh & 15;
  const int qt = blockIdx.x * 64;
  const int wv = threadIdx.x >> 5, lane = threadIdx.x & 31;
  const int hl = lane >> 4, lr = lane & 15;
  const int qRow0 = qt + wv * 16;

  const __bf16* qbase = q + (size_t)bh * L * Dh;
  const __bf16* kbase = k + (size_t)bh * L * Dh;
  const __bf16* vtbase = vt + (size_t)bh * Dh * L;

  // Q A-fragments for all of D (4 x K=32)
  v16bf qf[4];
  {
    const __bf16* qp = qbase + (size_t)(qRow0 + lr) * Dh;
#pragma unroll
    for (int kd = 0; kd < 4; ++kd) {
      const __bf16* p = qp + kd * 32 + hl * 8;
      qf[kd] = cat8(*(const v8bf*)p, *(const v8bf*)(p + 16));
    }
  }

  v8f acc[8] = {};
  float mrow[8], lsum[8];
#pragma unroll
  for (int j = 0; j < 8; ++j) { mrow[j] = -INFINITY; lsum[j] = 0.0f; }

  const int myLastRow = qRow0 + 15;
  const float sm_scale = 0.08838834764831845f;   // 1/sqrt(128)

  for (int kt = 0; kt < qt + 64; kt += 32) {
    __syncthreads();                   // previous tile fully consumed
    {   // stage K tile 32x128 (8 KB): 4 async b128 per thread
      const int r = threadIdx.x & 31, seg = threadIdx.x >> 5;
      const __bf16* src = kbase + (size_t)(kt + r) * Dh + seg * 32;
      __bf16* dst = &Ks[r][seg * 32];
#pragma unroll
      for (int i = 0; i < 4; ++i)
        async_copy_b128(dst + i * 8, src + i * 8);
    }
    {   // stage V^T tile 128x32 (8 KB): 4 async b128 per thread
      const int d = threadIdx.x;
      const __bf16* src = vtbase + (size_t)d * L + kt;
      __bf16* dst = &Vs[d][0];
#pragma unroll
      for (int i = 0; i < 4; ++i)
        async_copy_b128(dst + i * 8, src + i * 8);
    }
    wait_async0();                     // our ASYNCcnt isn't compiler-tracked
    __syncthreads();
    if (kt > myLastRow) continue;      // wave-uniform: EXEC stays all-ones

    // ---- scores S = Q K^T (16 x 32), 8 WMMA over D ----
    v8f s[2] = {};
#pragma unroll
    for (int kd = 0; kd < 4; ++kd)
#pragma unroll
      for (int ni = 0; ni < 2; ++ni) {
        const __bf16* p = &Ks[ni * 16 + lr][kd * 32 + hl * 8];
        const v16bf kf = cat8(*(const v8bf*)p, *(const v8bf*)(p + 16));
        s[ni] = wmma_bf16(qf[kd], kf, s[ni]);
      }

    // ---- scale + causal mask ----
#pragma unroll
    for (int ni = 0; ni < 2; ++ni) {
      const int key = kt + ni * 16 + lr;
#pragma unroll
      for (int j = 0; j < 8; ++j) {
        const int qr = qRow0 + j + hl * 8;
        const float sv = s[ni][j] * sm_scale;
        s[ni][j] = (key <= qr) ? sv : -INFINITY;
      }
    }

    // ---- online softmax (row = (j, half); width-16 butterflies) ----
#pragma unroll
    for (int j = 0; j < 8; ++j) {
      float mx = fmaxf(s[0][j], s[1][j]);
      mx = fmaxf(mx, __shfl_xor(mx, 1, 16));
      mx = fmaxf(mx, __shfl_xor(mx, 2, 16));
      mx = fmaxf(mx, __shfl_xor(mx, 4, 16));
      mx = fmaxf(mx, __shfl_xor(mx, 8, 16));
      const float mnew = fmaxf(mrow[j], mx);
      const float alpha = __expf(mrow[j] - mnew);
      mrow[j] = mnew;
      const float p0 = __expf(s[0][j] - mnew);
      const float p1 = __expf(s[1][j] - mnew);
      s[0][j] = p0; s[1][j] = p1;
      float rs = p0 + p1;
      rs += __shfl_xor(rs, 1, 16);
      rs += __shfl_xor(rs, 2, 16);
      rs += __shfl_xor(rs, 4, 16);
      rs += __shfl_xor(rs, 8, 16);
      lsum[j] = lsum[j] * alpha + rs;
#pragma unroll
      for (int d8 = 0; d8 < 8; ++d8) acc[d8][j] *= alpha;
    }

    // ---- P -> LDS (reshape D-layout -> A-fragment layout) ----
#pragma unroll
    for (int ni = 0; ni < 2; ++ni)
#pragma unroll
      for (int j = 0; j < 8; ++j)
        Ps[wv][j + hl * 8][ni * 16 + lr] = f2bf(s[ni][j]);
    asm volatile("s_wait_dscnt 0" ::: "memory");

    const __bf16* pp = &Ps[wv][lr][hl * 8];
    const v16bf pf = cat8(*(const v8bf*)pp, *(const v8bf*)(pp + 16));

    // ---- O += P V  (8 WMMA across D) ----
#pragma unroll
    for (int d8 = 0; d8 < 8; ++d8) {
      const __bf16* vp = &Vs[d8 * 16 + lr][hl * 8];
      const v16bf vf = cat8(*(const v8bf*)vp, *(const v8bf*)(vp + 16));
      acc[d8] = wmma_bf16(pf, vf, acc[d8]);
    }
  }

  // ---- normalize + store bf16 to [B, L, H*D] for output projection ----
#pragma unroll
  for (int j = 0; j < 8; ++j) {
    const float inv = 1.0f / lsum[j];
    const int qr = qRow0 + j + hl * 8;
    __bf16* dst = po + ((size_t)(b * L + qr) * 2048) + h * Dh;
#pragma unroll
    for (int d8 = 0; d8 < 8; ++d8)
      dst[d8 * 16 + lr] = f2bf(acc[d8][j] * inv);
  }
}

// ---------------------------------------------------------------------------
extern "C" void kernel_launch(void* const* d_in, const int* in_sizes, int n_in,
                              void* d_out, int out_size, void* d_ws,
                              size_t ws_size, hipStream_t stream) {
  (void)in_sizes; (void)n_in; (void)out_size; (void)ws_size;
  const float* x = (const float*)d_in[0];      // [2, 2048, 2048]
  const float* Wqkv = (const float*)d_in[1];   // [6144, 2048]
  const float* Wout = (const float*)d_in[2];   // [2048, 2048]
  float* out = (float*)d_out;                  // [2, 2048, 2048]

  constexpr int M = 4096;       // B*L
  constexpr int HID = 2048;

  char* ws = (char*)d_ws;
  float* qkv = (float*)ws;                                   // 96 MB f32
  size_t off = (size_t)M * 3 * HID * sizeof(float);
  __bf16* qb = (__bf16*)(ws + off);  off += (size_t)M * HID * 2;   // 16 MB
  __bf16* kb = (__bf16*)(ws + off);  off += (size_t)M * HID * 2;   // 16 MB
  __bf16* vtb = (__bf16*)(ws + off); off += (size_t)M * HID * 2;   // 16 MB
  __bf16* pob = (__bf16*)(ws + off);                               // 16 MB

  dim3 gQKV(M / 64, (3 * HID) / 128);
  gemm_bf16_wmma<false><<<gQKV, 256, 0, stream>>>(x, Wqkv, qkv, M, 3 * HID,
                                                  HID, 3 * HID);
  rmsnorm_rope_pack<<<M, 256, 0, stream>>>(qkv, qb, kb, vtb);
  dim3 gATT(2048 / 64, 32);
  attn_fwd_wmma<<<gATT, 128, 0, stream>>>(qb, kb, vtb, pob);
  dim3 gOUT(M / 64, HID / 128);
  gemm_bf16_wmma<true><<<gOUT, 256, 0, stream>>>(pob, Wout, out, M, HID, HID,
                                                 HID);
}
